// GCNModel_87402584474115
// MI455X (gfx1250) — matrix operand order repaired and checked by reading, not compiled, not run on hardware
//
#include <hip/hip_runtime.h>
#include <hip/hip_bf16.h>

// ---------------------------------------------------------------------------
// RGCN (basis decomposition) forward for MI455X / gfx1250, wave32 + WMMA bf16.
//   Y1[b]  = feat(f32->bf16) @ W1[b]      (WMMA 16x16x32 bf16, K=8192, split-K x4)
//   Xacc1  = sum_s A_s @ (Wc1[s,0]*Y1[0] + Wc1[s,1]*Y1[1])   (gather+atomic)
//   x1     = tanh(Xacc1) -> bf16
//   Y2[b]  = x1 @ W2[b]                    (WMMA, K=64)
//   Xacc2  = sum_s A_s @ (Wc2[s,0]*Y2[0] + Wc2[s,1]*Y2[1])
//   x2     = tanh(Xacc2);   out = x2 @ Wclf + bclf
//
// B-matrix tiles for the big GEMM are staged into LDS (double buffered,
// 16 KB per stage = 2 K-steps) via the Tensor Data Mover when available,
// else async global->LDS copies, else plain ds stores. All 8 waves then read
// fragments with batched ds loads; sched_group_barrier pins the per-K-step
// pipeline to {16 ds reads} -> {8 back-to-back WMMAs}.
// ---------------------------------------------------------------------------

typedef __bf16 bf16_t;
typedef __attribute__((ext_vector_type(16))) __bf16 v16bf;
typedef __attribute__((ext_vector_type(8)))  __bf16 v8bf;
typedef __attribute__((ext_vector_type(8)))  float  v8f;
typedef __attribute__((ext_vector_type(4)))  unsigned int u32x4;
typedef __attribute__((ext_vector_type(4)))  int i32x4;
typedef __attribute__((ext_vector_type(8)))  int i32x8;

#define WMMA_BF16(a, b, c) \
  __builtin_amdgcn_wmma_f32_16x16x32_bf16(false, (a), false, (b), (short)0, (c), false, false)

#define NN    8192
#define SS    4
#define BB    2
#define EE    262144      // 2^18
#define HH    64
#define FF    32
#define CC    2

// ---------------- CDNA5 feature detection ----------------
#if __has_builtin(__builtin_amdgcn_tensor_load_to_lds)
#define HAVE_TDM 1
#endif
#if __has_builtin(__builtin_amdgcn_global_load_async_to_lds_b128)
#define HAVE_ASYNC_LDS 1
#endif

#define AS1P(p) ((__attribute__((address_space(1))) void*)((void*)(p)))
#define AS3P(p) ((__attribute__((address_space(3))) void*)((void*)(p)))

// Pipeline shaping: {16 DS reads} then {8 WMMA} per K-step.
__device__ __forceinline__ void sched_pipeline_hint() {
#if __has_builtin(__builtin_amdgcn_sched_group_barrier)
  __builtin_amdgcn_sched_group_barrier(0x100, 16, 0);  // DS read group
  __builtin_amdgcn_sched_group_barrier(0x008, 8, 0);   // MFMA/WMMA group
#endif
}

__device__ __forceinline__ unsigned lds_byte_off(const void* p) {
  // generic -> LDS addrspace cast yields the in-LDS byte offset
  return (unsigned)(unsigned long long)(
      (__attribute__((address_space(3))) const void*)p);
}

__device__ __forceinline__ void wait_asynccnt0() {
#if __has_builtin(__builtin_amdgcn_s_wait_asynccnt)
  __builtin_amdgcn_s_wait_asynccnt(0);
#else
  asm volatile("s_wait_asynccnt 0" ::: "memory");
#endif
}

#if defined(HAVE_TDM)
// Issue one TDM descriptor copying 16 KB (8192 bf16, contiguous) -> LDS.
// D# group0: count=1, lds_addr, global_addr, type=2.
// D# group1: data_size=1(2B), tensor_dim0=8192, tensor_dim1=1, tile_dim0=8192,
//            tile_dim1/2 unused, tensor_dim0_stride=8192.
__device__ __forceinline__ void tdm_stage_b(const bf16_t* gsrc, unsigned ldsoff) {
  unsigned long long ga = (unsigned long long)(size_t)gsrc;
  u32x4 g0;
  g0.x = 1u;                                             // count=1, user desc
  g0.y = ldsoff;                                         // lds_addr (bytes)
  g0.z = (unsigned)(ga & 0xffffffffull);                 // global_addr[31:0]
  g0.w = (unsigned)((ga >> 32) & 0x01ffffffull) | (2u << 30);  // [56:32]+type=2
  i32x8 g1;
  g1[0] = 0x00010000;   // workgroup_mask=0, data_size=1 (2 bytes)
  g1[1] = 0x20000000;   // tensor_dim0[15:0]=8192 in bits[63:48]
  g1[2] = 0x00010000;   // tensor_dim0[31:16]=0, tensor_dim1[15:0]=1
  g1[3] = 0x20000000;   // tensor_dim1[31:16]=0, tile_dim0=8192
  g1[4] = 0;            // tile_dim1=0 (unused), tile_dim2=0
  g1[5] = 8192;         // tensor_dim0_stride[31:0]=8192
  g1[6] = 0;            // stride hi / dim1_stride lo
  g1[7] = 0;
  i32x4 z4 = {0, 0, 0, 0};
#if __clang_major__ >= 23
  i32x8 z8 = {0, 0, 0, 0, 0, 0, 0, 0};
  __builtin_amdgcn_tensor_load_to_lds(g0, g1, z4, z4, z8, 0);
#else
  __builtin_amdgcn_tensor_load_to_lds(g0, g1, z4, z4, 0);
#endif
}
#endif  // HAVE_TDM

// Cooperative stage of one 16 KB B tile into LDS (256 threads).
__device__ __forceinline__ void stage_b_tile(const bf16_t* __restrict__ gsrc,
                                             bf16_t* lbuf, int tid, int wave) {
#if defined(HAVE_TDM)
  (void)tid;
  if (wave == 0) tdm_stage_b(gsrc, lds_byte_off(lbuf));
#elif defined(HAVE_ASYNC_LDS)
  (void)wave;
  // each thread copies 64 bytes = 4 x b128
  const bf16_t* s = gsrc + tid * 32;
  bf16_t* d = lbuf + tid * 32;
#pragma unroll
  for (int j = 0; j < 4; ++j)
    __builtin_amdgcn_global_load_async_to_lds_b128(AS1P(s + j * 8),
                                                   AS3P(d + j * 8), 0, 0);
#else
  (void)wave;
  const uint4* s = (const uint4*)(gsrc + tid * 32);
  uint4* d = (uint4*)(lbuf + tid * 32);
#pragma unroll
  for (int j = 0; j < 4; ++j) d[j] = s[j];
#endif
}

__device__ __forceinline__ void stage_sync(int wave) {
#if defined(HAVE_TDM)
  if (wave == 0) __builtin_amdgcn_s_wait_tensorcnt((short)0);
  __syncthreads();
#elif defined(HAVE_ASYNC_LDS)
  wait_asynccnt0();
  __syncthreads();
#else
  __syncthreads();
#endif
}

// ---------------------------------------------------------------------------
// Pack W1 [B=2, 8192, 64] f32 -> bf16 B-matrix fragments.
// Fragment per (kt, ct): 32 lanes x 16 bf16 contiguous (1 KB).
// Lane L<16 : column N = ct*16+L,      K = kt*32 + {0..7, 16..23}
// Lane L>=16: column N = ct*16+(L-16), K = kt*32 + {8..15, 24..31}
// Columns 0..63 = base 0, 64..127 = base 1.
// ---------------------------------------------------------------------------
__global__ __launch_bounds__(256) void rgcn_pack_w1(const float* __restrict__ W1,
                                                    bf16_t* __restrict__ Wpk) {
  int t = blockIdx.x * 256 + threadIdx.x;       // 0 .. 8192*128-1
  int i    = t & 15;
  int lane = (t >> 4) & 31;
  int ct   = (t >> 9) & 7;
  int kt   = t >> 12;
  int col  = ct * 16 + (lane & 15);
  int b    = col >> 6;
  int h    = col & 63;
  int K    = kt * 32 + ((lane & 16) ? 8 : 0) + (i < 8 ? i : i + 8);
  Wpk[t] = (bf16_t)W1[((size_t)b * NN + K) * HH + h];
}

// Pack W2 [B=2, 64, 32] f32 -> bf16 fragments; columns 0..31 base0, 32..63 base1.
__global__ __launch_bounds__(256) void rgcn_pack_w2(const float* __restrict__ W2,
                                                    bf16_t* __restrict__ Wpk) {
  int t = blockIdx.x * 256 + threadIdx.x;       // 0 .. 4095
  int i    = t & 15;
  int lane = (t >> 4) & 31;
  int ct   = (t >> 9) & 3;
  int kt   = t >> 11;
  int col  = ct * 16 + (lane & 15);
  int b    = col >> 5;
  int f    = col & 31;
  int K    = kt * 32 + ((lane & 16) ? 8 : 0) + (i < 8 ? i : i + 8);
  Wpk[t] = (bf16_t)W2[((size_t)b * HH + K) * FF + f];
}

// ---------------------------------------------------------------------------
// GEMM1:  Y1[n, col] += sum_k feat[n,k] * W1pk[k, col],  col in [0,128).
// Grid (64, 4): blockIdx.x = M group (8 waves x 16 rows), blockIdx.y = K split
// (2048 K each = 32 staged pairs of K-steps). B tiles double-buffered in LDS.
// ---------------------------------------------------------------------------
__global__ __launch_bounds__(256) void rgcn_gemm1_wmma(const float*  __restrict__ feat,
                                                       const bf16_t* __restrict__ Bpk,
                                                       float*        __restrict__ Y1) {
  __shared__ __align__(32) bf16_t Bsh[2][8192];   // 2 x 16 KB

  const int tid   = threadIdx.x;
  const int lane  = tid & 31;
  const int wave  = tid >> 5;
  const int mtile = blockIdx.x * 8 + wave;        // 0..511
  const int m0    = mtile * 16;
  const int row   = m0 + (lane & 15);
  const int kshift = (lane & 16) ? 8 : 0;         // upper half-wave takes K+8
  const int kt0   = blockIdx.y * 64;              // 64 K-steps of 32 per split
  const int p0    = blockIdx.y * 32;              // staged pairs (2 kt each)

  v8f acc[8] = {};

  const float* arow = feat + (size_t)row * NN + (size_t)kt0 * 32 + kshift;

  // prologue: stage first 16 KB B tile
  stage_b_tile(Bpk + (size_t)p0 * 8192, &Bsh[0][0], tid, wave);
  stage_sync(wave);

  int buf = 0;
  for (int p = 0; p < 32; ++p) {
    if (p + 1 < 32)
      stage_b_tile(Bpk + (size_t)(p0 + p + 1) * 8192, &Bsh[buf ^ 1][0], tid, wave);

#pragma unroll
    for (int half = 0; half < 2; ++half) {
      const int kt = 2 * p + half;

      // ---- A fragment: fp32 loads -> bf16 cvt, per ISA 16-bit A layout ----
      const float4* ap = (const float4*)(arow + kt * 32);
      __builtin_prefetch(arow + (kt + 8) * 32, 0, 0);   // global_prefetch_b8
      float4 a0 = ap[0];   // K = base+0..3   (upper lanes: +8..)
      float4 a1 = ap[1];   // K = base+4..7
      float4 a2 = ap[4];   // K = base+16..19
      float4 a3 = ap[5];   // K = base+20..23

      v16bf afrag;
      afrag[0]  = (bf16_t)a0.x; afrag[1]  = (bf16_t)a0.y;
      afrag[2]  = (bf16_t)a0.z; afrag[3]  = (bf16_t)a0.w;
      afrag[4]  = (bf16_t)a1.x; afrag[5]  = (bf16_t)a1.y;
      afrag[6]  = (bf16_t)a1.z; afrag[7]  = (bf16_t)a1.w;
      afrag[8]  = (bf16_t)a2.x; afrag[9]  = (bf16_t)a2.y;
      afrag[10] = (bf16_t)a2.z; afrag[11] = (bf16_t)a2.w;
      afrag[12] = (bf16_t)a3.x; afrag[13] = (bf16_t)a3.y;
      afrag[14] = (bf16_t)a3.z; afrag[15] = (bf16_t)a3.w;

      // ---- batched B fragment reads (ds clause), then 8 WMMAs ----
      const bf16_t* bbase = &Bsh[buf][half * 4096 + lane * 16];
      v16bf bfr[8];
#pragma unroll
      for (int ct = 0; ct < 8; ++ct)
        bfr[ct] = *(const v16bf*)(bbase + ct * 512);
#pragma unroll
      for (int ct = 0; ct < 8; ++ct)
        acc[ct] = WMMA_BF16(afrag, bfr[ct], acc[ct]);

      sched_pipeline_hint();   // {16 ds reads} -> {8 wmma}
    }

    stage_sync(wave);
    buf ^= 1;
  }

  // C/D layout: lane<16 -> M=m0+r, N=lane ; lane>=16 -> M=m0+8+r, N=lane-16.
  // Split-K partials combine with fp32 atomics.
  const int mst = m0 + ((lane & 16) ? 8 : 0);
  const int nc  = lane & 15;
#pragma unroll
  for (int ct = 0; ct < 8; ++ct)
#pragma unroll
    for (int r = 0; r < 8; ++r)
      atomicAdd(Y1 + (size_t)(mst + r) * 128 + ct * 16 + nc, acc[ct][r]);
}

// ---------------------------------------------------------------------------
// GEMM2:  Y2[n, col] = sum_k x1bf[n,k] * W2pk[k, col], K=64, col in [0,64).
// ---------------------------------------------------------------------------
__global__ __launch_bounds__(256) void rgcn_gemm2_wmma(const bf16_t* __restrict__ x1bf,
                                                       const bf16_t* __restrict__ Bpk,
                                                       float*        __restrict__ Y2) {
  const int lane  = threadIdx.x & 31;
  const int wave  = threadIdx.x >> 5;
  const int mtile = blockIdx.x * 8 + wave;      // 0..511
  const int m0    = mtile * 16;
  const int row   = m0 + (lane & 15);
  const int kshift = (lane & 16) ? 8 : 0;

  v8f acc[4] = {};
  const bf16_t* xrow = x1bf + (size_t)row * HH + kshift;

#pragma unroll
  for (int kt = 0; kt < 2; ++kt) {
    v8bf lo = *(const v8bf*)(xrow + kt * 32);        // K base+0..7
    v8bf hi = *(const v8bf*)(xrow + kt * 32 + 16);   // K base+16..23
    v16bf afrag = __builtin_shufflevector(lo, hi, 0, 1, 2, 3, 4, 5, 6, 7,
                                          8, 9, 10, 11, 12, 13, 14, 15);
    const bf16_t* bbase = Bpk + (size_t)kt * 2048 + lane * 16;
    v16bf bfr[4];
#pragma unroll
    for (int ct = 0; ct < 4; ++ct)
      bfr[ct] = *(const v16bf*)(bbase + ct * 512);
#pragma unroll
    for (int ct = 0; ct < 4; ++ct)
      acc[ct] = WMMA_BF16(afrag, bfr[ct], acc[ct]);
  }

  const int mst = m0 + ((lane & 16) ? 8 : 0);
  const int nc  = lane & 15;
#pragma unroll
  for (int ct = 0; ct < 4; ++ct)
#pragma unroll
    for (int r = 0; r < 8; ++r)
      Y2[(size_t)(mst + r) * 64 + ct * 16 + nc] = acc[ct][r];
}

// ---------------------------------------------------------------------------
// Sparse aggregation, layer 1: thread = (edge, h), h in [0,64).
// ---------------------------------------------------------------------------
__global__ __launch_bounds__(256) void rgcn_agg1(const float* __restrict__ Y1,
                                                 const int*   __restrict__ esrc,
                                                 const int*   __restrict__ edst,
                                                 const float* __restrict__ ew,
                                                 const float* __restrict__ Wc,
                                                 float*       __restrict__ Xacc) {
  long long t = (long long)blockIdx.x * 256 + threadIdx.x;
  int h = (int)(t & 63);
  long long idx = t >> 6;                 // 0 .. S*E-1
  int s = (int)(idx >> 18);               // E = 2^18
  int src = esrc[idx];
  int dst = edst[idx];
  float w = ew[idx];
  float wc0 = Wc[s * 2 + 0];
  float wc1 = Wc[s * 2 + 1];
  float v = w * (wc0 * Y1[(size_t)src * 128 + h] +
                 wc1 * Y1[(size_t)src * 128 + 64 + h]);
  atomicAdd(Xacc + (size_t)dst * HH + h, v);
}

// Sparse aggregation, layer 2: h in [0,32).
__global__ __launch_bounds__(256) void rgcn_agg2(const float* __restrict__ Y2,
                                                 const int*   __restrict__ esrc,
                                                 const int*   __restrict__ edst,
                                                 const float* __restrict__ ew,
                                                 const float* __restrict__ Wc,
                                                 float*       __restrict__ Xacc) {
  long long t = (long long)blockIdx.x * 256 + threadIdx.x;
  int h = (int)(t & 31);
  long long idx = t >> 5;
  int s = (int)(idx >> 18);
  int src = esrc[idx];
  int dst = edst[idx];
  float w = ew[idx];
  float wc0 = Wc[s * 2 + 0];
  float wc1 = Wc[s * 2 + 1];
  float v = w * (wc0 * Y2[(size_t)src * 64 + h] +
                 wc1 * Y2[(size_t)src * 64 + 32 + h]);
  atomicAdd(Xacc + (size_t)dst * FF + h, v);
}

// tanh + cast to bf16 (layer-1 activation feeding WMMA GEMM2)
__global__ __launch_bounds__(256) void rgcn_tanh_bf16(const float* __restrict__ in,
                                                      bf16_t* __restrict__ out, int n) {
  int i = blockIdx.x * 256 + threadIdx.x;
  if (i < n) out[i] = (bf16_t)tanhf(in[i]);
}

// tanh, fp32 (layer-2 activation)
__global__ __launch_bounds__(256) void rgcn_tanh_f32(const float* __restrict__ in,
                                                     float* __restrict__ out, int n) {
  int i = blockIdx.x * 256 + threadIdx.x;
  if (i < n) out[i] = tanhf(in[i]);
}

// Classifier: out[n,c] = sum_f x2[n,f]*Wclf[f,c] + bclf[c]
__global__ __launch_bounds__(256) void rgcn_classify(const float* __restrict__ x2,
                                                     const float* __restrict__ Wclf,
                                                     const float* __restrict__ bclf,
                                                     float* __restrict__ out) {
  int n = blockIdx.x * 256 + threadIdx.x;
  if (n >= NN) return;
  float a0 = bclf[0], a1 = bclf[1];
#pragma unroll
  for (int f = 0; f < FF; ++f) {
    float v = x2[(size_t)n * FF + f];
    a0 += v * Wclf[f * CC + 0];
    a1 += v * Wclf[f * CC + 1];
  }
  out[(size_t)n * CC + 0] = a0;
  out[(size_t)n * CC + 1] = a1;
}

// ---------------------------------------------------------------------------
// Host-side launch
// ---------------------------------------------------------------------------
extern "C" void kernel_launch(void* const* d_in, const int* in_sizes, int n_in,
                              void* d_out, int out_size, void* d_ws, size_t ws_size,
                              hipStream_t stream) {
  const float* feat  = (const float*)d_in[0];   // [8192, 8192]
  const float* ew    = (const float*)d_in[1];   // [4, 262144]
  const float* W1    = (const float*)d_in[2];   // [2, 8192, 64]
  const float* Wc1   = (const float*)d_in[3];   // [4, 2]
  const float* W2    = (const float*)d_in[4];   // [2, 64, 32]
  const float* Wc2   = (const float*)d_in[5];   // [4, 2]
  const float* Wclf  = (const float*)d_in[6];   // [32, 2]
  const float* bclf  = (const float*)d_in[7];   // [2]
  const int*   esrc  = (const int*)d_in[8];     // [4, 262144]
  const int*   edst  = (const int*)d_in[9];     // [4, 262144]
  float*       out   = (float*)d_out;           // [8192, 2]

  char* ws = (char*)d_ws;
  bf16_t* Wpk1  = (bf16_t*)(ws);                        //  2 MB : W1 fragments
  float*  Y1    = (float*) (ws + (size_t)2  * 1048576); //  4 MB : [8192,128]
  float*  Xacc1 = (float*) (ws + (size_t)6  * 1048576); //  2 MB : [8192,64]
  bf16_t* x1bf  = (bf16_t*)(ws + (size_t)8  * 1048576); //  1 MB : [8192,64]
  bf16_t* Wpk2  = (bf16_t*)(ws + (size_t)9  * 1048576); //  8 KB : W2 fragments
  float*  Y2    = (float*) (ws + (size_t)10 * 1048576); //  2 MB : [8192,64]
  float*  Xacc2 = (float*) (ws + (size_t)12 * 1048576); //  1 MB : [8192,32]
  float*  X2    = (float*) (ws + (size_t)13 * 1048576); //  1 MB : [8192,32]

  // Zero atomic accumulators (graph-capture-safe async memsets).
  hipMemsetAsync(Y1,    0, (size_t)NN * 128 * sizeof(float), stream);
  hipMemsetAsync(Xacc1, 0, (size_t)NN * HH  * sizeof(float), stream);
  hipMemsetAsync(Xacc2, 0, (size_t)NN * FF  * sizeof(float), stream);

  // Pack weights into bf16 WMMA B-fragments.
  rgcn_pack_w1<<<4096, 256, 0, stream>>>(W1, Wpk1);
  rgcn_pack_w2<<<16,   256, 0, stream>>>(W2, Wpk2);

  // Layer 1 dense: Y1 = feat @ [W1[0] | W1[1]]  (split-K x4)
  rgcn_gemm1_wmma<<<dim3(64, 4), 256, 0, stream>>>(feat, Wpk1, Y1);

  // Layer 1 sparse aggregation + tanh.
  rgcn_agg1<<<(int)(((long long)SS * EE * 64) / 256), 256, 0, stream>>>(
      Y1, esrc, edst, ew, Wc1, Xacc1);
  rgcn_tanh_bf16<<<(NN * HH + 255) / 256, 256, 0, stream>>>(Xacc1, x1bf, NN * HH);

  // Layer 2 dense: Y2 = x1 @ [W2[0] | W2[1]]
  rgcn_gemm2_wmma<<<64, 256, 0, stream>>>(x1bf, Wpk2, Y2);

  // Layer 2 sparse aggregation + tanh.
  rgcn_agg2<<<(int)(((long long)SS * EE * 32) / 256), 256, 0, stream>>>(
      Y2, esrc, edst, ew, Wc2, Xacc2);
  rgcn_tanh_f32<<<(NN * FF + 255) / 256, 256, 0, stream>>>(Xacc2, X2, NN * FF);

  // Classifier.
  rgcn_classify<<<NN / 256, 256, 0, stream>>>(X2, Wclf, bclf, out);
}
